// StructuralEquationModels_2267742732424
// MI455X (gfx1250) — matrix operand order repaired and checked by reading, not compiled, not run on hardware
//
#include <hip/hip_runtime.h>
#include <hip/hip_bf16.h>
#include <cmath>

typedef __bf16 bf16;
typedef bf16 v16bf __attribute__((ext_vector_type(16)));
typedef bf16 v8bf  __attribute__((ext_vector_type(8)));
typedef bf16 v2bf  __attribute__((ext_vector_type(2)));
typedef float v8f  __attribute__((ext_vector_type(8)));

#define WMMA(a, b, c) \
  __builtin_amdgcn_wmma_f32_16x16x32_bf16(false, (a), false, (b), (short)0, (c), false, false)

// Problem constants
#define DNUM   32      // number of equations
#define VD     64      // var_dim
#define HID    256     // hidden
#define INDIM  2048    // d * var_dim
#define BM     128     // batch tile rows per block

// LDS strides (bf16 elements unless noted); every fragment row base is 16B aligned
#define SA   72        // A tile: 128 x 72 bf16  (rows 144B = 16*9)
#define SB   72        // B tile (transposed W1 chunk): 256 x 72 bf16
#define SH   264       // H tile: 128 x 264 bf16 (rows 528B = 16*33)
#define SW   264       // W2^T: 64 x 264 bf16
#define SY   68        // Y tile: 128 x 68 f32 (rows 272B = 16*17)

// LDS layout (bytes)
#define OFF_A     0                      // A tile: 18432
#define OFF_B0    18432                  // B ping buffer: 36864
#define B_BYTES   (256 * SB * 2)         // 36864 per B buffer
#define OFF_AUX   (OFF_B0 + 2 * B_BYTES) // 92160: W2^T (33792) / Y (34816)
#define LDS_TOTAL (OFF_AUX + BM * SY * 4)   // 92160 + 34816 = 126976

// A-fragment (16x32 bf16, M x K): lane L holds row M=L%16.
// L<16: K = kc+{0..7} and kc+{16..23}; L>=16: +8. Pairs packed per VGPR.
__device__ __forceinline__ v16bf load_fragA(const bf16* base, int row, int stride,
                                            int kc, int lane) {
  const bf16* p = base + (row + (lane & 15)) * stride + kc + ((lane >> 4) & 1) * 8;
  v8bf lo = *(const v8bf*)(p);
  v8bf hi = *(const v8bf*)(p + 16);
  v16bf r;
#pragma unroll
  for (int q = 0; q < 8; ++q) { r[q] = lo[q]; r[q + 8] = hi[q]; }
  return r;
}

// B-fragment (32x16 bf16, K x N) from K-contiguous (transposed) storage:
// lane L holds column N=L%16; L<16: K = kc+{0..7} and kc+{16..23}; L>=16: +8.
__device__ __forceinline__ v16bf load_fragB(const bf16* base, int row, int stride,
                                            int kc, int lane) {
  const bf16* p = base + (row + (lane & 15)) * stride + kc + ((lane >> 4) & 1) * 8;
  v8bf lo = *(const v8bf*)(p);
  v8bf hi = *(const v8bf*)(p + 16);
  v16bf r;
#pragma unroll
  for (int q = 0; q < 8; ++q) { r[q] = lo[q]; r[q + 8] = hi[q]; }
  return r;
}

__global__ void __launch_bounds__(256)
sem_fused_kernel(const float* __restrict__ cvars,  // (B, d, 64)
                 const float* __restrict__ adj,    // (d, d)
                 const float* __restrict__ W1,     // (d, 2048, 256)
                 const float* __restrict__ b1,     // (d, 256)
                 const float* __restrict__ W2,     // (d, 256, 64)
                 const float* __restrict__ b2,     // (d, 64)
                 const float* __restrict__ gamma,  // (d, 64)
                 const float* __restrict__ beta,   // (d, 64)
                 float* __restrict__ out)          // (B, d, 64)
{
  extern __shared__ __align__(16) char smem[];
  bf16*  ldsA  = (bf16*)(smem + OFF_A);
  bf16*  ldsB0 = (bf16*)(smem + OFF_B0);        // ping; pong = ldsB0 + 256*SB
  bf16*  ldsH  = (bf16*)smem;                   // aliases A+B region after GEMM1
  bf16*  ldsW2 = (bf16*)(smem + OFF_AUX);
  float* ldsY  = (float*)(smem + OFF_AUX);      // aliases W2^T after GEMM2

  const int i    = blockIdx.x;        // equation index
  const int b0   = blockIdx.y * BM;   // batch tile start
  const int t    = threadIdx.x;
  const int lane = t & 31;
  const int wave = t >> 5;
  const int wm   = wave & 3;          // 4 wave-rows  (32 output rows each)
  const int wn   = wave >> 2;         // 2 wave-cols

  const float* W1i = W1 + (size_t)i * INDIM * HID;
  const float* W2i = W2 + (size_t)i * HID * VD;

  // ---- Stage W2^T (bf16) into LDS once: ldsW2[n][k] ----
#pragma unroll 1
  for (int it = 0; it < 32; ++it) {
    int idx = it * 256 + t;
    int kp  = idx >> 6;               // 0..127 -> k = 2*kp
    int n   = idx & 63;
    float f0 = W2i[(size_t)(2 * kp) * VD + n];
    float f1 = W2i[(size_t)(2 * kp + 1) * VD + n];
    v2bf pk = {(bf16)f0, (bf16)f1};
    *(v2bf*)&ldsW2[n * SW + 2 * kp] = pk;
  }

  // Staging helpers (per thread): B tile half h (kp = h*16+q) of chunk jj.
  auto issueB = [&](int jj, int h, float2* bR) {
#pragma unroll
    for (int q = 0; q < 16; ++q) {
      const float* src = W1i + (size_t)(jj * 64 + 2 * (h * 16 + q)) * HID + t;
      bR[q].x = src[0];
      bR[q].y = src[HID];
    }
  };
  auto storeB = [&](bf16* dstB, int h, const float2* bR) {
#pragma unroll
    for (int q = 0; q < 16; ++q) {
      v2bf pk = {(bf16)bR[q].x, (bf16)bR[q].y};
      *(v2bf*)&dstB[t * SB + 2 * (h * 16 + q)] = pk;
    }
  };

  // ---- Prologue: stage B chunk j=0 into ping buffer ----
  {
    float2 bR[16];
    issueB(0, 0, bR); storeB(ldsB0, 0, bR);
    issueB(0, 1, bR); storeB(ldsB0, 1, bR);
  }

  // =========================== GEMM1 ===========================
  // H[128 x 256] = gelu( (adj-scaled c)[128 x 2048] * W1_i[2048 x 256] + b1 )
  v8f acc[2][8] = {};

  // One kc-step: 16 WMMAs with a 1-deep pipeline over the 8 B-fragments.
  auto compute_kc = [&](const bf16* Bc, int kc) {
    v16bf a0 = load_fragA(ldsA, wm * 32 + 0,  SA, kc, lane);
    v16bf a1 = load_fragA(ldsA, wm * 32 + 16, SA, kc, lane);
    v16bf bb = load_fragB(Bc, wn * 128, SB, kc, lane);
#pragma unroll
    for (int nf = 0; nf < 8; ++nf) {
      int nfn = (nf < 7) ? nf + 1 : 7;
      v16bf bnext = load_fragB(Bc, wn * 128 + nfn * 16, SB, kc, lane);
      acc[0][nf] = WMMA(a0, bb, acc[0][nf]);
      acc[1][nf] = WMMA(a1, bb, acc[1][nf]);
      bb = bnext;
    }
  };

#pragma unroll 1
  for (int j = 0; j < DNUM; ++j) {
    const int bc = j & 1;
    const int jn = (j + 1) & (DNUM - 1);   // wraps on last iter (harmless dead stage)
    bf16* Bcur = ldsB0 + bc * (256 * SB);
    bf16* Bnxt = ldsB0 + (bc ^ 1) * (256 * SB);

    // Stage A(j): c[b0:b0+128, j, :] * adj[i][j] -> ldsA (L2-hot, single-buffered)
    {
      const float aij = adj[i * DNUM + j];
      int r0 = t >> 5;                // 0..7
      int vp = lane * 2;              // even column
#pragma unroll
      for (int pass = 0; pass < 16; ++pass) {
        int r = pass * 8 + r0;
        const float* src = cvars + ((size_t)(b0 + r) * DNUM + j) * VD + vp;
        float f0 = src[0] * aij, f1 = src[1] * aij;
        v2bf pk = {(bf16)f0, (bf16)f1};
        *(v2bf*)&ldsA[r * SA + vp] = pk;
      }
    }

    // Prefetch next W1 chunk, half 0, into registers (in flight during compute)
    float2 bR[16];
    issueB(jn, 0, bR);

    __syncthreads();   // A(j) visible; B(j) (stored last iter) visible

    compute_kc(Bcur, 0);

    storeB(Bnxt, 0, bR);     // Bnxt readers finished before end-of-prev-iter barrier
    issueB(jn, 1, bR);       // half 1 in flight during second kc step

    compute_kc(Bcur, 32);

    storeB(Bnxt, 1, bR);
    __syncthreads();   // A(j) consumers done before A(j+1) overwrite; B(j+1) stores done
  }

  // ---- bias + exact gelu -> bf16 H in LDS (overwrites A/B region) ----
  {
    const float* b1i = b1 + i * HID;
    const int colbase = wn * 128 + (lane & 15);
    const int rowoff  = wm * 32 + ((lane >> 4) & 1) * 8;
#pragma unroll
    for (int nf = 0; nf < 8; ++nf) {
      int col = colbase + nf * 16;
      float bias = b1i[col];
#pragma unroll
      for (int mf = 0; mf < 2; ++mf) {
#pragma unroll
        for (int r = 0; r < 8; ++r) {
          float x = acc[mf][nf][r] + bias;
          float g = 0.5f * x * (1.0f + erff(x * 0.70710678118654752f));
          ldsH[(rowoff + mf * 16 + r) * SH + col] = (bf16)g;
        }
      }
    }
  }
  __syncthreads();

  // =========================== GEMM2 ===========================
  // Y[128 x 64] = H[128 x 256] * W2_i[256 x 64]; wave tile 32x32
  v8f acc2[2][2] = {};
#pragma unroll 1
  for (int kb = 0; kb < 8; ++kb) {
    int kc = kb * 32;
    v16bf a0  = load_fragA(ldsH,  wm * 32 + 0,  SH, kc, lane);
    v16bf a1  = load_fragA(ldsH,  wm * 32 + 16, SH, kc, lane);
    v16bf bb0 = load_fragB(ldsW2, wn * 32 + 0,  SW, kc, lane);
    v16bf bb1 = load_fragB(ldsW2, wn * 32 + 16, SW, kc, lane);
    acc2[0][0] = WMMA(a0, bb0, acc2[0][0]);
    acc2[0][1] = WMMA(a0, bb1, acc2[0][1]);
    acc2[1][0] = WMMA(a1, bb0, acc2[1][0]);
    acc2[1][1] = WMMA(a1, bb1, acc2[1][1]);
  }
  __syncthreads();  // done reading ldsH/ldsW2; Y overwrites W2 region

  // ---- y + b2 -> f32 Y tile in LDS ----
  {
    const float* b2i = b2 + i * VD;
    const int rowoff = wm * 32 + ((lane >> 4) & 1) * 8;
#pragma unroll
    for (int nf = 0; nf < 2; ++nf) {
      int col = wn * 32 + nf * 16 + (lane & 15);
      float bias = b2i[col];
#pragma unroll
      for (int mf = 0; mf < 2; ++mf) {
#pragma unroll
        for (int r = 0; r < 8; ++r) {
          ldsY[(rowoff + mf * 16 + r) * SY + col] = acc2[mf][nf][r] + bias;
        }
      }
    }
  }
  __syncthreads();

  // ---- LayerNorm over var_dim=64, one thread per batch row ----
  if (t < BM) {
    const float* yr = ldsY + t * SY;
    float s = 0.f, s2 = 0.f;
#pragma unroll
    for (int v = 0; v < VD; ++v) { float y = yr[v]; s += y; s2 += y * y; }
    float mu  = s * (1.0f / VD);
    float var = s2 * (1.0f / VD) - mu * mu;
    float rs  = rsqrtf(var + 1e-5f);
    const float* gi = gamma + i * VD;
    const float* bi = beta + i * VD;
    float* o = out + ((size_t)(b0 + t) * DNUM + i) * VD;
#pragma unroll
    for (int v = 0; v < VD; v += 4) {
      float4 yv = *(const float4*)(yr + v);
      float4 g  = *(const float4*)(gi + v);
      float4 be = *(const float4*)(bi + v);
      float4 r;
      r.x = (yv.x - mu) * rs * g.x + be.x;
      r.y = (yv.y - mu) * rs * g.y + be.y;
      r.z = (yv.z - mu) * rs * g.z + be.z;
      r.w = (yv.w - mu) * rs * g.w + be.w;
      *(float4*)(o + v) = r;
    }
  }
}

extern "C" void kernel_launch(void* const* d_in, const int* in_sizes, int n_in,
                              void* d_out, int out_size, void* d_ws, size_t ws_size,
                              hipStream_t stream) {
  const float* cvars = (const float*)d_in[0];  // (1024, 32, 64)
  const float* adj   = (const float*)d_in[1];  // (32, 32)
  const float* W1    = (const float*)d_in[2];  // (32, 2048, 256)
  const float* b1    = (const float*)d_in[3];  // (32, 256)
  const float* W2    = (const float*)d_in[4];  // (32, 256, 64)
  const float* b2    = (const float*)d_in[5];  // (32, 64)
  const float* gam   = (const float*)d_in[6];  // (32, 64)
  const float* bet   = (const float*)d_in[7];  // (32, 64)
  float* out = (float*)d_out;                  // (1024, 32, 64)

  dim3 grid(DNUM, 1024 / BM);   // 32 equations x 8 batch tiles = 256 workgroups
  dim3 block(256);              // 8 waves (wave32)
  sem_fused_kernel<<<grid, block, LDS_TOTAL, stream>>>(
      cvars, adj, W1, b1, W2, b2, gam, bet, out);
}
